// AutoEncoder_75282186764670
// MI455X (gfx1250) — compile-verified, matrix operand order checked by reading, not run on hardware
//
#include <hip/hip_runtime.h>
#include <hip/hip_fp16.h>

#define Nn   131072
#define Bn   32768
#define DIMn 128
#define HIDn 64
#define MAXPn 17
#define Rn   (Bn * MAXPn)   // 557056, divisible by 16

typedef __attribute__((ext_vector_type(16))) _Float16 v16h;
typedef __attribute__((ext_vector_type(8)))  _Float16 v8h;
typedef __attribute__((ext_vector_type(8)))  float    v8f;

#define WMMA16(a,b,c) __builtin_amdgcn_wmma_f32_16x16x32_f16(false,(a),false,(b),(short)0,(c),false,false)

// ---- fragment helpers -------------------------------------------------------
// A (16x32 f16): lane L holds row M=L%16; per kb the lane's halves are two
// contiguous 8-half chunks at K = kbase + 8*(L>=16) and K = kbase+16+8*(L>=16).
__device__ __forceinline__ v16h frag_a_load(const _Float16* p, int laneHi) {
  union { v16h v; v8h h[2]; } u;
  u.h[0] = *(const v8h*)(p + 8 * laneHi);
  u.h[1] = *(const v8h*)(p + 16 + 8 * laneHi);
  return u.v;
}
// B fragments are pre-packed: packed[((ct*kbT+kb)*32+lane)*16 + e]
__device__ __forceinline__ v16h frag_b_load(const _Float16* packed, int kbT, int ct, int kb, int lane) {
  return *(const v16h*)(packed + (((size_t)ct * kbT + kb) * 32 + lane) * 16);
}

// ---- pack a [K0,N0] f32 row-major weight into B-fragment layout (zero pad) --
__global__ void pack_b_kernel(const float* __restrict__ W, int K0, int N0,
                              int kbT, int ctT, _Float16* __restrict__ out) {
  int idx = blockIdx.x * blockDim.x + threadIdx.x;
  int total = ctT * kbT * 32 * 16;
  if (idx >= total) return;
  int e = idx & 15;
  int lane = (idx >> 4) & 31;
  int kb = (idx >> 9) % kbT;
  int ct = idx / (kbT * 512);
  int laneHi = lane >> 4;
  int n = ct * 16 + (lane & 15);
  int k = kb * 32 + ((e < 8) ? (8 * laneHi + e) : (16 + 8 * laneHi + (e - 8)));
  float v = (k < K0 && n < N0) ? W[(size_t)k * N0 + n] : 0.0f;
  out[idx] = (_Float16)v;
}

// ---- positional tables: MLP_LN(one_hot(r)) (row rows-1 = zero input if zero_last)
__global__ void postab_kernel(const float* __restrict__ w1, const float* __restrict__ b1,
                              const float* __restrict__ g, const float* __restrict__ bt,
                              const float* __restrict__ w2, const float* __restrict__ b2,
                              float* __restrict__ out, int rows, int zero_last) {
  int r = threadIdx.x;
  if (r >= rows) return;
  bool zerorow = zero_last && (r == rows - 1);
  float h[40];
  for (int j = 0; j < 40; ++j) h[j] = b1[j] + (zerorow ? 0.f : w1[r * 40 + j]);
  float m = 0.f;
  for (int j = 0; j < 40; ++j) m += h[j];
  m *= (1.f / 40.f);
  float v = 0.f;
  for (int j = 0; j < 40; ++j) { float d = h[j] - m; v += d * d; }
  v *= (1.f / 40.f);
  float rs = rsqrtf(v + 1e-5f);
  for (int j = 0; j < 40; ++j) h[j] = tanhf((h[j] - m) * rs * g[j] + bt[j]);
  for (int o = 0; o < 64; ++o) {
    float s = b2[o];
    for (int j = 0; j < 40; ++j) s += h[j] * w2[j * 64 + o];
    out[r * 64 + o] = s;
  }
}

// ---- rank scores + f16 copy of x -------------------------------------------
__global__ void rank_kernel(const float* __restrict__ x, const float* __restrict__ rank_w,
                            const float* __restrict__ rank_b,
                            float* __restrict__ mag, _Float16* __restrict__ xh) {
  int i = blockIdx.x * blockDim.x + threadIdx.x;
  if (i >= Nn) return;
  const float* xr = x + (size_t)i * DIMn;
  float s = 0.f;
  for (int t = 0; t < DIMn; ++t) {
    float v = xr[t];
    s += v * rank_w[t];
    xh[(size_t)i * DIMn + t] = (_Float16)v;
  }
  mag[i] = s + rank_b[0];
}

// ---- per-set start/count via binary search (batch is sorted) ----------------
__global__ void seg_kernel(const int* __restrict__ batch, int* __restrict__ segstart,
                           int* __restrict__ nvals) {
  int b = blockIdx.x * blockDim.x + threadIdx.x;
  if (b >= Bn) return;
  int lo = 0, hi = Nn;
  while (lo < hi) { int mid = (lo + hi) >> 1; if (batch[mid] < b) lo = mid + 1; else hi = mid; }
  int s = lo;
  lo = s; hi = Nn;
  while (lo < hi) { int mid = (lo + hi) >> 1; if (batch[mid] <= b) lo = mid + 1; else hi = mid; }
  segstart[b] = s;
  nvals[b] = lo - s;
}

// ---- stable rank of each element within its set (matches lexsort) -----------
__global__ void key_kernel(const int* __restrict__ batch, const int* __restrict__ segstart,
                           const int* __restrict__ nvals, const float* __restrict__ mag,
                           int* __restrict__ key) {
  int i = blockIdx.x * blockDim.x + threadIdx.x;
  if (i >= Nn) return;
  int b = batch[i];
  int s = segstart[b], e = s + nvals[b];
  float mi = mag[i];
  int r = 0;
  for (int j = s; j < e; ++j) {
    float mj = mag[j];
    r += (mj < mi) || (mj == mi && j < i);
  }
  key[i] = r;
}

#define NWAVE 4

// ---- encoder psi: [N,128] -> 96 (LN, tanh) -> 64  (WMMA f16 in, f32 acc) ----
__global__ __launch_bounds__(128) void psi_kernel(
    const _Float16* __restrict__ xh,
    const _Float16* __restrict__ w1p, const float* __restrict__ b1,
    const float* __restrict__ g, const float* __restrict__ bt,
    const _Float16* __restrict__ w2p, const float* __restrict__ b2,
    float* __restrict__ psi) {
  __shared__ __align__(16) float    h1[NWAVE][16][96];
  __shared__ __align__(16) _Float16 h1h[NWAVE][16][96];
  const int lane = threadIdx.x & 31;
  const int wv = threadIdx.x >> 5;
  const int tile = blockIdx.x * NWAVE + wv;
  const int laneHi = lane >> 4;
  const int col = lane & 15;
  const int row0 = tile * 16;

  for (int ct = 0; ct < 6; ++ct) {
    v8f acc = {};
#pragma unroll
    for (int kb = 0; kb < 4; ++kb) {
      v16h a = frag_a_load(xh + (size_t)(row0 + col) * DIMn + kb * 32, laneHi);
      v16h b = frag_b_load(w1p, 4, ct, kb, lane);
      acc = WMMA16(a, b, acc);
    }
    float bias = b1[ct * 16 + col];
#pragma unroll
    for (int r = 0; r < 8; ++r)
      h1[wv][r + 8 * laneHi][ct * 16 + col] = acc[r] + bias;
  }
  __syncthreads();
  if (lane < 16) {
    float m = 0.f;
    for (int t = 0; t < 96; ++t) m += h1[wv][lane][t];
    m *= (1.f / 96.f);
    float v = 0.f;
    for (int t = 0; t < 96; ++t) { float d = h1[wv][lane][t] - m; v += d * d; }
    v *= (1.f / 96.f);
    float rs = rsqrtf(v + 1e-5f);
    for (int t = 0; t < 96; ++t)
      h1h[wv][lane][t] = (_Float16)tanhf((h1[wv][lane][t] - m) * rs * g[t] + bt[t]);
  }
  __syncthreads();
  for (int ct = 0; ct < 4; ++ct) {
    v8f acc = {};
#pragma unroll
    for (int kb = 0; kb < 3; ++kb) {
      v16h a = frag_a_load(&h1h[wv][col][0] + kb * 32, laneHi);
      v16h b = frag_b_load(w2p, 3, ct, kb, lane);
      acc = WMMA16(a, b, acc);
    }
    float bias = b2[ct * 16 + col];
#pragma unroll
    for (int r = 0; r < 8; ++r)
      psi[(size_t)(row0 + r + 8 * laneHi) * HIDn + ct * 16 + col] = acc[r] + bias;
  }
}

// ---- segment sum of psi_i * pee[rank_i] -> y2 (f16 for next GEMM) ----------
__global__ void y2_kernel(const float* __restrict__ psi, const int* __restrict__ segstart,
                          const int* __restrict__ nvals, const int* __restrict__ key,
                          const float* __restrict__ pee, _Float16* __restrict__ y2h) {
  int lane = threadIdx.x & 31;
  int b = blockIdx.x * (blockDim.x >> 5) + (threadIdx.x >> 5);
  if (b >= Bn) return;
  int s = segstart[b], n = nvals[b];
  float a0 = 0.f, a1 = 0.f;
  for (int t = 0; t < n; ++t) {
    int i = s + t;
    int k = key[i]; if (k > 17) k = 17;
    a0 += psi[(size_t)i * HIDn + lane] * pee[k * 64 + lane];
    a1 += psi[(size_t)i * HIDn + lane + 32] * pee[k * 64 + lane + 32];
  }
  y2h[(size_t)b * HIDn + lane] = (_Float16)a0;
  y2h[(size_t)b * HIDn + lane + 32] = (_Float16)a1;
}

// ---- phi: [B,64(+onehot row)] -> 72 (tanh) -> 64 ----------------------------
__global__ __launch_bounds__(128) void phi_kernel(
    const _Float16* __restrict__ y2h,
    const _Float16* __restrict__ w1p, const float* __restrict__ phi_w1,
    const float* __restrict__ b1,
    const _Float16* __restrict__ w2p, const float* __restrict__ b2,
    const int* __restrict__ nvals, float* __restrict__ z) {
  __shared__ __align__(16) _Float16 h1h[NWAVE][16][96];
  const int lane = threadIdx.x & 31;
  const int wv = threadIdx.x >> 5;
  const int tile = blockIdx.x * NWAVE + wv;
  const int laneHi = lane >> 4;
  const int col = lane & 15;
  const int row0 = tile * 16;

  if (lane < 16)
    for (int t = 80; t < 96; ++t) h1h[wv][lane][t] = (_Float16)0.f;

  for (int ct = 0; ct < 5; ++ct) {
    v8f acc = {};
#pragma unroll
    for (int kb = 0; kb < 2; ++kb) {
      v16h a = frag_a_load(y2h + (size_t)(row0 + col) * HIDn + kb * 32, laneHi);
      v16h b = frag_b_load(w1p, 2, ct, kb, lane);
      acc = WMMA16(a, b, acc);
    }
    int c = ct * 16 + col;
#pragma unroll
    for (int r = 0; r < 8; ++r) {
      int rr = r + 8 * laneHi;
      float val = 0.f;
      if (c < 72) {
        int nb = nvals[row0 + rr];
        val = acc[r] + b1[c];
        if (nb < 17) val += phi_w1[(size_t)(64 + nb) * 72 + c];  // one-hot concat fold
        val = tanhf(val);
      }
      h1h[wv][rr][c] = (_Float16)val;
    }
  }
  __syncthreads();
  for (int ct = 0; ct < 4; ++ct) {
    v8f acc = {};
#pragma unroll
    for (int kb = 0; kb < 3; ++kb) {
      v16h a = frag_a_load(&h1h[wv][col][0] + kb * 32, laneHi);
      v16h b = frag_b_load(w2p, 3, ct, kb, lane);
      acc = WMMA16(a, b, acc);
    }
    float bias = b2[ct * 16 + col];
#pragma unroll
    for (int r = 0; r < 8; ++r)
      z[(size_t)(row0 + r + 8 * laneHi) * HIDn + ct * 16 + col] = acc[r] + bias;
  }
}

// ---- size predictor + argmax -----------------------------------------------
__global__ void size_kernel(const float* __restrict__ z, const float* __restrict__ w1,
                            const float* __restrict__ b1, const float* __restrict__ w2,
                            const float* __restrict__ b2, int* __restrict__ ndec) {
  int b = blockIdx.x * blockDim.x + threadIdx.x;
  if (b >= Bn) return;
  const float* zr = z + (size_t)b * HIDn;
  float h[40];
  for (int j = 0; j < 40; ++j) {
    float s = b1[j];
    for (int k = 0; k < HIDn; ++k) s += zr[k] * w1[k * 40 + j];
    h[j] = tanhf(s);
  }
  int best = 0; float bestv = -3.4e38f;
  for (int o = 0; o < MAXPn; ++o) {
    float s = b2[o];
    for (int j = 0; j < 40; ++j) s += h[j] * w2[j * MAXPn + o];
    if (s > bestv) { bestv = s; best = o; }
  }
  ndec[b] = best;
}

// ---- decoder: zp=z*pe -> 96 (tanh) -> 128, masked ---------------------------
__global__ __launch_bounds__(128) void dec_kernel(
    const float* __restrict__ z, const float* __restrict__ ped, const int* __restrict__ ndec,
    const _Float16* __restrict__ w1p, const float* __restrict__ b1,
    const _Float16* __restrict__ w2p, const float* __restrict__ b2,
    float* __restrict__ xr) {
  __shared__ __align__(16) _Float16 a0[NWAVE][16][64];
  __shared__ __align__(16) _Float16 h1h[NWAVE][16][96];
  __shared__ float pedS[MAXPn * 64];
  for (int t = threadIdx.x; t < MAXPn * 64; t += blockDim.x) pedS[t] = ped[t];
  __syncthreads();

  const int lane = threadIdx.x & 31;
  const int wv = threadIdx.x >> 5;
  const int tile = blockIdx.x * NWAVE + wv;
  const int laneHi = lane >> 4;
  const int col = lane & 15;
  const int row0 = tile * 16;

  for (int j = 0; j < 16; ++j) {
    int gr = row0 + j;
    int b = gr / MAXPn, p = gr % MAXPn;
    a0[wv][j][lane]      = (_Float16)(z[(size_t)b * HIDn + lane]      * pedS[p * 64 + lane]);
    a0[wv][j][lane + 32] = (_Float16)(z[(size_t)b * HIDn + lane + 32] * pedS[p * 64 + lane + 32]);
  }
  __syncthreads();
  for (int ct = 0; ct < 6; ++ct) {
    v8f acc = {};
#pragma unroll
    for (int kb = 0; kb < 2; ++kb) {
      v16h a = frag_a_load(&a0[wv][col][0] + kb * 32, laneHi);
      v16h b = frag_b_load(w1p, 2, ct, kb, lane);
      acc = WMMA16(a, b, acc);
    }
    float bias = b1[ct * 16 + col];
#pragma unroll
    for (int r = 0; r < 8; ++r)
      h1h[wv][r + 8 * laneHi][ct * 16 + col] = (_Float16)tanhf(acc[r] + bias);
  }
  __syncthreads();
  int keep[8];
#pragma unroll
  for (int r = 0; r < 8; ++r) {
    int gr = row0 + r + 8 * laneHi;
    keep[r] = (gr % MAXPn) < ndec[gr / MAXPn];
  }
  for (int ct = 0; ct < 8; ++ct) {
    v8f acc = {};
#pragma unroll
    for (int kb = 0; kb < 3; ++kb) {
      v16h a = frag_a_load(&h1h[wv][col][0] + kb * 32, laneHi);
      v16h b = frag_b_load(w2p, 3, ct, kb, lane);
      acc = WMMA16(a, b, acc);
    }
    float bias = b2[ct * 16 + col];
#pragma unroll
    for (int r = 0; r < 8; ++r) {
      int gr = row0 + r + 8 * laneHi;
      xr[(size_t)gr * DIMn + ct * 16 + col] = keep[r] ? (acc[r] + bias) : 0.f;
    }
  }
}

// ---- batchr / mask outputs --------------------------------------------------
__global__ void meta_kernel(const int* __restrict__ ndec, float* __restrict__ outB,
                            float* __restrict__ outM) {
  int r = blockIdx.x * blockDim.x + threadIdx.x;
  if (r >= Rn) return;
  int b = r / MAXPn, p = r % MAXPn;
  outB[r] = (float)b;
  outM[r] = (p < ndec[b]) ? 1.f : 0.f;
}

// ---- host side --------------------------------------------------------------
extern "C" void kernel_launch(void* const* d_in, const int* in_sizes, int n_in,
                              void* d_out, int out_size, void* d_ws, size_t ws_size,
                              hipStream_t stream) {
  const float* x       = (const float*)d_in[0];
  const int*   batch   = (const int*)d_in[1];
  const float* rank_w  = (const float*)d_in[2];
  const float* rank_b  = (const float*)d_in[3];
  const float* psi_w1  = (const float*)d_in[4];
  const float* psi_b1  = (const float*)d_in[5];
  const float* psi_g   = (const float*)d_in[6];
  const float* psi_bt  = (const float*)d_in[7];
  const float* psi_w2  = (const float*)d_in[8];
  const float* psi_b2  = (const float*)d_in[9];
  const float* pe_e_w1 = (const float*)d_in[10];
  const float* pe_e_b1 = (const float*)d_in[11];
  const float* pe_e_g  = (const float*)d_in[12];
  const float* pe_e_bt = (const float*)d_in[13];
  const float* pe_e_w2 = (const float*)d_in[14];
  const float* pe_e_b2 = (const float*)d_in[15];
  const float* phi_w1  = (const float*)d_in[16];
  const float* phi_b1  = (const float*)d_in[17];
  const float* phi_w2  = (const float*)d_in[18];
  const float* phi_b2  = (const float*)d_in[19];
  const float* size_w1 = (const float*)d_in[20];
  const float* size_b1 = (const float*)d_in[21];
  const float* size_w2 = (const float*)d_in[22];
  const float* size_b2 = (const float*)d_in[23];
  const float* pe_d_w1 = (const float*)d_in[24];
  const float* pe_d_b1 = (const float*)d_in[25];
  const float* pe_d_g  = (const float*)d_in[26];
  const float* pe_d_bt = (const float*)d_in[27];
  const float* pe_d_w2 = (const float*)d_in[28];
  const float* pe_d_b2 = (const float*)d_in[29];
  const float* dec_w1  = (const float*)d_in[30];
  const float* dec_b1  = (const float*)d_in[31];
  const float* dec_w2  = (const float*)d_in[32];
  const float* dec_b2  = (const float*)d_in[33];
  (void)in_sizes; (void)n_in; (void)out_size; (void)ws_size;

  char* ws = (char*)d_ws;
  size_t off = 0;
  auto alloc = [&](size_t bytes) { size_t o = off; off = (off + bytes + 255) & ~(size_t)255; return o; };

  _Float16* xh   = (_Float16*)(ws + alloc((size_t)Nn * DIMn * 2));
  float*    mag  = (float*)(ws + alloc((size_t)Nn * 4));
  int*      segs = (int*)(ws + alloc((size_t)Bn * 4));
  int*      nval = (int*)(ws + alloc((size_t)Bn * 4));
  int*      key  = (int*)(ws + alloc((size_t)Nn * 4));
  float*    psi  = (float*)(ws + alloc((size_t)Nn * HIDn * 4));
  _Float16* y2h  = (_Float16*)(ws + alloc((size_t)Bn * HIDn * 2));
  float*    z    = (float*)(ws + alloc((size_t)Bn * HIDn * 4));
  int*      ndec = (int*)(ws + alloc((size_t)Bn * 4));
  float*    pee  = (float*)(ws + alloc(18 * 64 * 4));
  float*    pedt = (float*)(ws + alloc(17 * 64 * 4));
  _Float16* psiW1p = (_Float16*)(ws + alloc((size_t)6 * 4 * 512 * 2));
  _Float16* psiW2p = (_Float16*)(ws + alloc((size_t)4 * 3 * 512 * 2));
  _Float16* phiW1p = (_Float16*)(ws + alloc((size_t)5 * 2 * 512 * 2));
  _Float16* phiW2p = (_Float16*)(ws + alloc((size_t)4 * 3 * 512 * 2));
  _Float16* decW1p = (_Float16*)(ws + alloc((size_t)6 * 2 * 512 * 2));
  _Float16* decW2p = (_Float16*)(ws + alloc((size_t)8 * 3 * 512 * 2));

  float* xrOut = (float*)d_out;
  float* bOut  = xrOut + (size_t)Rn * DIMn;
  float* mOut  = bOut + Rn;

  // weight repack (B-fragment layout, zero padded)
  pack_b_kernel<<<(6 * 4 * 512 + 255) / 256, 256, 0, stream>>>(psi_w1, 128, 96, 4, 6, psiW1p);
  pack_b_kernel<<<(4 * 3 * 512 + 255) / 256, 256, 0, stream>>>(psi_w2, 96, 64, 3, 4, psiW2p);
  pack_b_kernel<<<(5 * 2 * 512 + 255) / 256, 256, 0, stream>>>(phi_w1, 64, 72, 2, 5, phiW1p);
  pack_b_kernel<<<(4 * 3 * 512 + 255) / 256, 256, 0, stream>>>(phi_w2, 72, 64, 3, 4, phiW2p);
  pack_b_kernel<<<(6 * 2 * 512 + 255) / 256, 256, 0, stream>>>(dec_w1, 64, 96, 2, 6, decW1p);
  pack_b_kernel<<<(8 * 3 * 512 + 255) / 256, 256, 0, stream>>>(dec_w2, 96, 128, 3, 8, decW2p);
  // positional tables (row 17 of pee = zero-input overflow case)
  postab_kernel<<<1, 32, 0, stream>>>(pe_e_w1, pe_e_b1, pe_e_g, pe_e_bt, pe_e_w2, pe_e_b2, pee, 18, 1);
  postab_kernel<<<1, 32, 0, stream>>>(pe_d_w1, pe_d_b1, pe_d_g, pe_d_bt, pe_d_w2, pe_d_b2, pedt, 17, 0);

  rank_kernel<<<Nn / 256, 256, 0, stream>>>(x, rank_w, rank_b, mag, xh);
  seg_kernel<<<Bn / 256, 256, 0, stream>>>(batch, segs, nval);
  key_kernel<<<Nn / 256, 256, 0, stream>>>(batch, segs, nval, mag, key);

  psi_kernel<<<(Nn / 16) / NWAVE, 128, 0, stream>>>(xh, psiW1p, psi_b1, psi_g, psi_bt,
                                                    psiW2p, psi_b2, psi);
  y2_kernel<<<Bn / 4, 128, 0, stream>>>(psi, segs, nval, key, pee, y2h);
  phi_kernel<<<(Bn / 16) / NWAVE, 128, 0, stream>>>(y2h, phiW1p, phi_w1, phi_b1,
                                                    phiW2p, phi_b2, nval, z);
  size_kernel<<<Bn / 256, 256, 0, stream>>>(z, size_w1, size_b1, size_w2, size_b2, ndec);
  dec_kernel<<<(Rn / 16) / NWAVE, 128, 0, stream>>>(z, pedt, ndec, decW1p, dec_b1,
                                                    decW2p, dec_b2, xrOut);
  meta_kernel<<<(Rn + 255) / 256, 256, 0, stream>>>(ndec, bOut, mOut);
}
